// AGAEMD_30794915512681
// MI455X (gfx1250) — matrix operand order repaired and checked by reading, not compiled, not run on hardware
//
#include <hip/hip_runtime.h>
#include <stdint.h>

// ---------------------------------------------------------------------------
// AGAEMD 3-layer GAT + h@h^T for MI455X (gfx1250, wave32, WMMA f16->f32).
// Compute-bound (AI > 2000 FLOP/B): all GEMMs on v_wmma_f32_16x16x32_f16.
// Aggregation (90% of FLOPs) streams B tiles with GLOBAL_LOAD_ASYNC_TO_LDS
// (ASYNCcnt) + double buffering: DMA of tile k+1 overlaps WMMA on tile k,
// one barrier per iteration. Attention tiles synthesized in registers from
// per-row softmax stats + bit-packed adjacency (2MB, L2-resident).
// ---------------------------------------------------------------------------

typedef _Float16 f16;
typedef __attribute__((ext_vector_type(16))) _Float16 v16h;
typedef __attribute__((ext_vector_type(8)))  _Float16 h8;
typedef __attribute__((ext_vector_type(8)))  float    v8f;

#define N_NODES 4096
#define FDIM    256
#define HEADS   4
#define SLOPE   0.2f

// ---- CDNA5 async tensor-data path -----------------------------------------
// global -> LDS DMA, tracked by ASYNCcnt (CDNA5 ISA).
// LDS operand is a VGPR holding the byte offset from the workgroup LDS base;
// generic pointers to __shared__ carry that offset in addr[31:0].
__device__ __forceinline__ void async_ld_b128(f16* lds_dst, const f16* gsrc) {
  unsigned lds_off = (unsigned)(uintptr_t)lds_dst;
  asm volatile("global_load_async_to_lds_b128 %0, %1, off"
               :: "v"(lds_off), "v"(gsrc) : "memory");
}
__device__ __forceinline__ void wait_async0() {
  asm volatile("s_wait_asynccnt 0x0" ::: "memory");
}

// ---- WMMA fragment loaders (wave32 layouts per CDNA5 ISA 7.12.2) ----------
// A 16x32 f16, row-major tile [16][32] in LDS:
//   lane (r=lane&15, hi=lane>>4): a[e] = tile[r][(e>>3)*16 + hi*8 + (e&7)]
__device__ __forceinline__ v16h frag_a_rowmajor(const f16* tile, int lane) {
  const int r = lane & 15, hi = (lane >> 4) & 1;
  h8 x0 = *(const h8*)(tile + r * 32 + hi * 8);
  h8 x1 = *(const h8*)(tile + r * 32 + 16 + hi * 8);
  v16h out;
#pragma unroll
  for (int e = 0; e < 8; ++e) { out[e] = x0[e]; out[e + 8] = x1[e]; }
  return out;
}

// B 32x16 f16 from its N-major transpose tileT[16 n][32 k] in LDS:
//   b[e] = tileT[n][(e>>3)*16 + hi*8 + (e&7)]
__device__ __forceinline__ v16h frag_b_ntile(const f16* tileT, int lane) {
  const int r = lane & 15, hi = (lane >> 4) & 1;
  h8 x0 = *(const h8*)(tileT + r * 32 + hi * 8);
  h8 x1 = *(const h8*)(tileT + r * 32 + 16 + hi * 8);
  v16h out;
#pragma unroll
  for (int e = 0; e < 8; ++e) { out[e] = x0[e]; out[e + 8] = x1[e]; }
  return out;
}

// ---- setup kernels --------------------------------------------------------
__global__ void k_cvt_x(const float* __restrict__ x, f16* __restrict__ xo) {
  int tid = blockIdx.x * 256 + threadIdx.x;
  xo[tid] = (f16)x[tid];
}

// W[h][f][g] f32 -> wT[h][g][f] f16 (B-transpose for the projection GEMM)
__global__ void k_cvt_w(const float* __restrict__ Wm, f16* __restrict__ wT) {
  int tid = blockIdx.x * 256 + threadIdx.x;   // tid = h*65536 + g*256 + f
  int h = tid >> 16, rem = tid & 65535, g = rem >> 8, f = rem & 255;
  wT[tid] = (f16)Wm[h * 65536 + f * 256 + g];
}

// adj bool(u8) [N][N] -> bit-packed [N][N/32] (16MB -> 2MB, lives in L2)
__global__ void k_pack(const unsigned char* __restrict__ adj,
                       unsigned* __restrict__ bits) {
  int tid = blockIdx.x * 256 + threadIdx.x;   // tid = i*128 + word
  const unsigned* pu = (const unsigned*)(adj + (size_t)tid * 32);
  unsigned b = 0;
#pragma unroll
  for (int k = 0; k < 8; ++k) {
    unsigned v = pu[k];
#pragma unroll
    for (int c = 0; c < 4; ++c)
      if ((v >> (8 * c)) & 0xFFu) b |= 1u << (k * 4 + c);
  }
  bits[tid] = b;
}

// ---- projection: hT[h][g][n] = (xin @ W[h])^T, WMMA tiled -----------------
// block = 128 thr (4 waves); block tile 64n x 64g; K loop over F in 32s.
__global__ __launch_bounds__(128)
void k_proj(const f16* __restrict__ xin, const f16* __restrict__ wT,
            f16* __restrict__ hT) {
  __shared__ f16 At[64 * 32];
  __shared__ f16 Bt[64 * 32];
  __shared__ f16 Tr[64 * 64];
  const int t = threadIdx.x;
  const int w = t >> 5, lane = t & 31, r16 = lane & 15, hi = (lane >> 4) & 1;
  const int g0 = blockIdx.x * 64, n0 = blockIdx.y * 64, h = blockIdx.z;
  const int rr = t >> 1, cc = (t & 1) * 16;
  v8f acc[4] = {};
  for (int f0 = 0; f0 < FDIM; f0 += 32) {
    __syncthreads();
    { const uint4* s  = (const uint4*)(xin + (n0 + rr) * FDIM + f0 + cc);
      uint4* d  = (uint4*)(At + rr * 32 + cc);  d[0] = s[0];  d[1] = s[1];
      const uint4* s2 = (const uint4*)(wT + (h * FDIM + g0 + rr) * FDIM + f0 + cc);
      uint4* d2 = (uint4*)(Bt + rr * 32 + cc);  d2[0] = s2[0]; d2[1] = s2[1]; }
    __syncthreads();
    v16h a = frag_a_rowmajor(At + w * 16 * 32, lane);
#pragma unroll
    for (int ct = 0; ct < 4; ++ct) {
      v16h b = frag_b_ntile(Bt + ct * 16 * 32, lane);
      acc[ct] = __builtin_amdgcn_wmma_f32_16x16x32_f16(
          false, a, false, b, (short)0, acc[ct], false, false);
    }
  }
  // transpose through LDS so the g-major global store is coalesced
  __syncthreads();
#pragma unroll
  for (int ct = 0; ct < 4; ++ct) {
    h8 tmp;
#pragma unroll
    for (int e = 0; e < 8; ++e) tmp[e] = (f16)acc[ct][e];
    // D elem e: n_local = w*16 + hi*8 + e, g_local = ct*16 + r16
    *(h8*)(Tr + (ct * 16 + r16) * 64 + w * 16 + hi * 8) = tmp;
  }
  __syncthreads();
  { const int gl = t >> 1, c = (t & 1) * 32;
    const uint4* s = (const uint4*)(Tr + gl * 64 + c);
    uint4* d = (uint4*)(hT + (h * FDIM + g0 + gl) * N_NODES + n0 + c);
    d[0] = s[0]; d[1] = s[1]; d[2] = s[2]; d[3] = s[3]; }
}

// ---- per-(head,node) attention coefficients f_src/f_dst -------------------
__global__ __launch_bounds__(256)
void k_dots(const f16* __restrict__ hT, const float* __restrict__ av,
            float* __restrict__ fsrc, float* __restrict__ fdst) {
  int tid = blockIdx.x * 256 + threadIdx.x;   // tid = h*N + n
  int h = tid >> 12, n = tid & (N_NODES - 1);
  float s0 = 0.f, s1 = 0.f;
  for (int g = 0; g < FDIM; ++g) {            // coalesced across n
    float v = (float)hT[(h * FDIM + g) * N_NODES + n];
    s0 += v * av[(h * FDIM + g) * 2 + 0];
    s1 += v * av[(h * FDIM + g) * 2 + 1];
  }
  fsrc[tid] = s0; fdst[tid] = s1;
}

// ---- per-row masked-softmax stats: m_i and 1/l_i (one wave per (h,i)) -----
__global__ __launch_bounds__(256)
void k_stats(const unsigned* __restrict__ bits, const float* __restrict__ fsrc,
             const float* __restrict__ fdst, float* __restrict__ mrow,
             float* __restrict__ invl) {
  int wid = blockIdx.x * 8 + (threadIdx.x >> 5);   // wid = h*N + i
  int lane = threadIdx.x & 31;
  int h = wid >> 12, i = wid & (N_NODES - 1);
  float fs = fsrc[h * N_NODES + i];
  const float* fd = fdst + h * N_NODES;
  float m = -3.0e38f, l = 0.f;                     // per-lane online softmax
  for (int wi = lane; wi < 128; wi += 32) {
    unsigned wv = bits[i * 128 + wi];
    while (wv) {
      int b = __ffs(wv) - 1;
      wv &= wv - 1;
      float s = fs + fd[wi * 32 + b];
      s = s > 0.f ? s : SLOPE * s;                 // leaky_relu
      if (s <= m) l += __expf(s - m);
      else { l = l * __expf(m - s) + 1.f; m = s; }
    }
  }
#pragma unroll
  for (int o = 16; o > 0; o >>= 1) {               // lane-tree (m,l) merge
    float om = __shfl_xor(m, o, 32);
    float ol = __shfl_xor(l, o, 32);
    float nm = fmaxf(m, om);
    float w1 = (m  > -1.0e30f) ? __expf(m  - nm) : 0.f;
    float w2 = (om > -1.0e30f) ? __expf(om - nm) : 0.f;
    l = l * w1 + ol * w2;
    m = nm;
  }
  if (lane == 0) { mrow[h * N_NODES + i] = m; invl[h * N_NODES + i] = 1.f / l; }
}

// ---- aggregation: out = mean_h elu(att_h @ h_h + xin), WMMA ---------------
// block = 128 thr (4 waves); block tile 64 rows x 128 feats. B tiles are
// DMA'd global->LDS (async, double buffered); att A-tiles synthesized in
// registers per 32-wide j block; single barrier per iteration.
__global__ __launch_bounds__(128)
void k_agg(const f16* __restrict__ hT, const f16* __restrict__ xin,
           const float* __restrict__ fsrc, const float* __restrict__ fdst,
           const float* __restrict__ mrow, const float* __restrict__ invl,
           const unsigned* __restrict__ bits, f16* __restrict__ xout) {
  __shared__ f16 Bt[2][128 * 32];
  const int t = threadIdx.x;
  const int w = t >> 5, lane = t & 31, r16 = lane & 15, hi = (lane >> 4) & 1;
  const int g0 = blockIdx.x * 128;
  const int n0 = blockIdx.y * 64 + w * 16;   // this wave's 16 output rows
  const int i  = n0 + r16;                   // lane's att row (A layout)
  v8f macc[8] = {};
  for (int h = 0; h < HEADS; ++h) {
    const float fs = fsrc[h * N_NODES + i];
    const float mi = mrow[h * N_NODES + i];
    const float il = invl[h * N_NODES + i];
    const float* fd = fdst + h * N_NODES;
    const f16* hsrc = hT + (size_t)(h * FDIM + g0 + t) * N_NODES;
    v8f acc[8] = {};
    // prologue: DMA tile 0 into buf0 (last-read buffer of the previous head
    // is always buf1, and its readers are past the final barrier -> safe)
    { f16* d = &Bt[0][t * 32];
      async_ld_b128(d + 0,  hsrc + 0);
      async_ld_b128(d + 8,  hsrc + 8);
      async_ld_b128(d + 16, hsrc + 16);
      async_ld_b128(d + 24, hsrc + 24); }
    int buf = 0;
    for (int jb = 0; jb < N_NODES; jb += 32, buf ^= 1) {
      wait_async0();            // my DMA batch for Bt[buf] landed
      __syncthreads();          // everyone's landed; prev buffer fully read
      if (jb + 32 < N_NODES) {  // DMA next tile while we compute this one
        const f16* s = hsrc + jb + 32;
        f16* d = &Bt[buf ^ 1][t * 32];
        async_ld_b128(d + 0,  s + 0);
        async_ld_b128(d + 8,  s + 8);
        async_ld_b128(d + 16, s + 16);
        async_ld_b128(d + 24, s + 24);
      }
      // --- build att A-fragment in registers ---
      unsigned word = bits[i * 128 + (jb >> 5)];
      const float4* fd4 = (const float4*)(fd + jb);
      float4 q0 = fd4[hi * 2], q1 = fd4[hi * 2 + 1];
      float4 q2 = fd4[4 + hi * 2], q3 = fd4[4 + hi * 2 + 1];
      float fdv[16] = {q0.x, q0.y, q0.z, q0.w, q1.x, q1.y, q1.z, q1.w,
                       q2.x, q2.y, q2.z, q2.w, q3.x, q3.y, q3.z, q3.w};
      v16h afrag;
#pragma unroll
      for (int e = 0; e < 16; ++e) {
        int jr = (e >> 3) * 16 + hi * 8 + (e & 7);    // j - jb
        float s = fs + fdv[e];
        s = s > 0.f ? s : SLOPE * s;
        float att = ((word >> jr) & 1u) ? __expf(s - mi) * il : 0.f;
        afrag[e] = (f16)att;
      }
#pragma unroll
      for (int ct = 0; ct < 8; ++ct) {
        v16h b = frag_b_ntile(&Bt[buf][ct * 16 * 32], lane);
        acc[ct] = __builtin_amdgcn_wmma_f32_16x16x32_f16(
            false, afrag, false, b, (short)0, acc[ct], false, false);
      }
    }
    // residual + ELU, accumulate head mean
#pragma unroll
    for (int ct = 0; ct < 8; ++ct)
#pragma unroll
      for (int e = 0; e < 8; ++e) {
        int row = n0 + hi * 8 + e;
        int col = g0 + ct * 16 + r16;
        float v = acc[ct][e] + (float)xin[row * FDIM + col];
        v = v > 0.f ? v : (__expf(v) - 1.f);
        macc[ct][e] += 0.25f * v;
      }
  }
#pragma unroll
  for (int ct = 0; ct < 8; ++ct)
#pragma unroll
    for (int e = 0; e < 8; ++e)
      xout[(n0 + hi * 8 + e) * FDIM + g0 + ct * 16 + r16] = (f16)macc[ct][e];
}

// ---- final: out[i][j] = sum_g x3[i][g]*x3[j][g], WMMA ---------------------
__global__ __launch_bounds__(128)
void k_final(const f16* __restrict__ x3, float* __restrict__ out) {
  __shared__ f16 At[64 * 32];
  __shared__ f16 Bt[64 * 32];
  const int t = threadIdx.x;
  const int w = t >> 5, lane = t & 31, r16 = lane & 15, hi = (lane >> 4) & 1;
  const int j0 = blockIdx.x * 64, i0 = blockIdx.y * 64;
  const int rr = t >> 1, cc = (t & 1) * 16;
  v8f acc[4] = {};
  for (int g0 = 0; g0 < FDIM; g0 += 32) {
    __syncthreads();
    { const uint4* s  = (const uint4*)(x3 + (i0 + rr) * FDIM + g0 + cc);
      uint4* d  = (uint4*)(At + rr * 32 + cc);  d[0] = s[0];  d[1] = s[1];
      const uint4* s2 = (const uint4*)(x3 + (j0 + rr) * FDIM + g0 + cc);
      uint4* d2 = (uint4*)(Bt + rr * 32 + cc);  d2[0] = s2[0]; d2[1] = s2[1]; }
    __syncthreads();
    v16h a = frag_a_rowmajor(At + w * 16 * 32, lane);
#pragma unroll
    for (int ct = 0; ct < 4; ++ct) {
      v16h b = frag_b_ntile(Bt + ct * 16 * 32, lane);
      acc[ct] = __builtin_amdgcn_wmma_f32_16x16x32_f16(
          false, a, false, b, (short)0, acc[ct], false, false);
    }
  }
#pragma unroll
  for (int ct = 0; ct < 4; ++ct)
#pragma unroll
    for (int e = 0; e < 8; ++e)
      out[(size_t)(i0 + w * 16 + hi * 8 + e) * N_NODES + j0 + ct * 16 + r16] =
          acc[ct][e];
}

// ---------------------------------------------------------------------------
extern "C" void kernel_launch(void* const* d_in, const int* in_sizes, int n_in,
                              void* d_out, int out_size, void* d_ws, size_t ws_size,
                              hipStream_t stream) {
  const float* x            = (const float*)d_in[0];
  const unsigned char* adj  = (const unsigned char*)d_in[1];  // bool -> u8
  const float* Wm           = (const float*)d_in[2];
  const float* av           = (const float*)d_in[3];

  char* ws = (char*)d_ws;
  size_t o = 0;
  f16* xA = (f16*)(ws + o); o += (size_t)N_NODES * FDIM * 2;         // 2MB
  f16* xB = (f16*)(ws + o); o += (size_t)N_NODES * FDIM * 2;         // 2MB
  f16* wT = (f16*)(ws + o); o += (size_t)HEADS * FDIM * FDIM * 2;    // 512KB
  f16* hT = (f16*)(ws + o); o += (size_t)HEADS * FDIM * N_NODES * 2; // 8MB
  float* fs = (float*)(ws + o); o += (size_t)HEADS * N_NODES * 4;
  float* fd = (float*)(ws + o); o += (size_t)HEADS * N_NODES * 4;
  float* mr = (float*)(ws + o); o += (size_t)HEADS * N_NODES * 4;
  float* il = (float*)(ws + o); o += (size_t)HEADS * N_NODES * 4;
  unsigned* bits = (unsigned*)(ws + o); o += (size_t)N_NODES * N_NODES / 8;

  k_cvt_x<<<(N_NODES * FDIM) / 256, 256, 0, stream>>>(x, xA);
  k_cvt_w<<<(HEADS * FDIM * FDIM) / 256, 256, 0, stream>>>(Wm, wT);
  k_pack<<<(N_NODES * 128) / 256, 256, 0, stream>>>(adj, bits);

  f16* cur = xA; f16* nxt = xB;
  for (int L = 0; L < 3; ++L) {
    k_proj<<<dim3(FDIM / 64, N_NODES / 64, HEADS), 128, 0, stream>>>(cur, wT, hT);
    k_dots<<<(HEADS * N_NODES) / 256, 256, 0, stream>>>(hT, av, fs, fd);
    k_stats<<<(HEADS * N_NODES) / 8, 256, 0, stream>>>(bits, fs, fd, mr, il);
    k_agg<<<dim3(FDIM / 128, N_NODES / 64), 128, 0, stream>>>(hT, cur, fs, fd,
                                                              mr, il, bits, nxt);
    f16* tmp = cur; cur = nxt; nxt = tmp;
  }
  k_final<<<dim3(N_NODES / 64, N_NODES / 64), 128, 0, stream>>>(cur, (float*)d_out);

  (void)in_sizes; (void)n_in; (void)out_size; (void)ws_size;
}
